// Attention_43542378446922
// MI455X (gfx1250) — matrix operand order repaired and checked by reading, not compiled, not run on hardware
//
#include <hip/hip_runtime.h>
#include <hip/hip_bf16.h>

typedef __attribute__((ext_vector_type(16))) _Float16 v16h;
typedef __attribute__((ext_vector_type(8)))  _Float16 v8h;
typedef __attribute__((ext_vector_type(8)))  float    v8f;

#define B_   2
#define T_   4096
#define C_   768
#define H_   12
#define D_   64
#define BT_  (B_ * T_)          // 8192
#define N3C_ (3 * C_)           // 2304

__device__ __forceinline__ v8f wmma16(v16h a, v16h b, v8f c) {
  return __builtin_amdgcn_wmma_f32_16x16x32_f16(
      /*neg_a=*/false, a, /*neg_b=*/false, b,
      /*c_mod=*/(short)0, c, /*reuse_a=*/false, /*reuse_b=*/false);
}

// ---- async copy: 16 bytes per lane, global -> LDS (ASYNCcnt-tracked) ----
__device__ __forceinline__ void async_b128_to_lds(const void* g, void* l) {
  unsigned lds = (unsigned)(unsigned long long)l;   // low 32 bits = LDS offset
  unsigned long long ga = (unsigned long long)g;
  asm volatile("global_load_async_to_lds_b128 %0, %1, off"
               :: "v"(lds), "v"(ga) : "memory");
}
__device__ __forceinline__ void wait_async_le(int more) {
  if (more) asm volatile("s_wait_asynccnt 0x1" ::: "memory");
  else      asm volatile("s_wait_asynccnt 0x0" ::: "memory");
}
__device__ __forceinline__ void wait_async_le4(int more) {
  if (more) asm volatile("s_wait_asynccnt 0x4" ::: "memory");
  else      asm volatile("s_wait_asynccnt 0x0" ::: "memory");
}

// A fragment (16x32 f16, row-major source, leading dim ld in elements).
__device__ __forceinline__ v16h load_a_frag(const _Float16* __restrict__ base,
                                            int ld, int row0, int k0) {
  int lane = threadIdx.x & 31;
  int r    = row0 + (lane & 15);
  int koff = k0 + ((lane >> 4) << 3);
  const _Float16* p = base + (size_t)r * ld + koff;
  v8h lo = *(const v8h*)p;
  v8h hi = *(const v8h*)(p + 16);
  v16h a;
#pragma unroll
  for (int i = 0; i < 8; ++i) { a[i] = lo[i]; a[8 + i] = hi[i]; }
  return a;
}

// Build B fragment from a row-major [K][N] tile: lane l = column l%16, K in table order.
__device__ __forceinline__ v16h pack_b(const _Float16* p, int pitch) {
  int lane = threadIdx.x & 31;
  const _Float16* q = p + (lane & 15) + ((lane >> 4) << 3) * pitch;
  v16h b;
#pragma unroll
  for (int i = 0; i < 8; ++i) { b[i] = q[i * pitch]; b[8 + i] = q[(16 + i) * pitch]; }
  return b;
}

__global__ void cvt_f32_f16(const float* __restrict__ in,
                            _Float16* __restrict__ out, int n) {
  int i = blockIdx.x * blockDim.x + threadIdx.x;
  if (i < n) out[i] = (_Float16)in[i];
}

// ---------------- shared GEMM body: B tile staged in LDS, double-buffered ----
#define BK 32
#define BN 64
#define BPITCH 72   // halves; 144B row pitch

__device__ __forceinline__ void gemm_body(const _Float16* __restrict__ A, int lda,
                                          const _Float16* __restrict__ Bm, int ldb,
                                          int m0, int n0,
                                          _Float16 (&ldsB)[2][BK][BPITCH],
                                          v8f (&acc)[4]) {
  const int tid  = threadIdx.x;
  const int lane = tid & 31;
  const int nk   = C_ / BK;                 // 24
  const int lr   = tid >> 3;                // 0..31: B row this thread copies
  const int lc   = (tid & 7) * 8;           // 16B chunk within row

  async_b128_to_lds(Bm + (size_t)lr * ldb + n0 + lc, &ldsB[0][lr][lc]);
  for (int i = 0; i < nk; ++i) {
    int k0 = i * BK;
    int more = (i + 1 < nk);
    if (more)
      async_b128_to_lds(Bm + (size_t)(k0 + BK + lr) * ldb + n0 + lc,
                        &ldsB[(i + 1) & 1][lr][lc]);
    wait_async_le(more);
    __syncthreads();
    __builtin_prefetch(A + (size_t)(m0 + (lane & 15)) * lda + k0 + 64, 0, 1);
    v16h a = load_a_frag(A, lda, m0, k0);
    const _Float16 (*bt)[BPITCH] = ldsB[i & 1];
#pragma unroll
    for (int j = 0; j < 4; ++j)
      acc[j] = wmma16(a, pack_b(&bt[0][16 * j], BPITCH), acc[j]);
    __syncthreads();
  }
}

// qkv = xh[8192x768] @ wh[768x2304]; scatter into Qh[bh,t,d], Kt[bh,d,t], Vh[bh,t,d]
__global__ void qkv_gemm(const _Float16* __restrict__ xh,
                         const _Float16* __restrict__ wh,
                         _Float16* __restrict__ Qh,
                         _Float16* __restrict__ Kt,
                         _Float16* __restrict__ Vh) {
  __shared__ _Float16 ldsB[2][BK][BPITCH];
  int wave = threadIdx.x >> 5;
  int lane = threadIdx.x & 31;
  int m0 = blockIdx.x * 128 + wave * 16;
  int n0 = blockIdx.y * 64;
  v8f acc[4] = {v8f{}, v8f{}, v8f{}, v8f{}};
  gemm_body(xh, C_, wh, N3C_, m0, n0, ldsB, acc);

  int mbase = (lane >> 4) * 8;
  int ncol  = lane & 15;
#pragma unroll
  for (int j = 0; j < 4; ++j) {
#pragma unroll
    for (int g = 0; g < 8; ++g) {
      int row = m0 + g + mbase;
      int n   = n0 + 16 * j + ncol;
      int region = n / C_;
      int hd = n % C_;
      int h = hd >> 6, d = hd & 63;
      int b = row >> 12, t = row & (T_ - 1);
      int bh = b * H_ + h;
      _Float16 hv = (_Float16)acc[j][g];
      if (region == 0)      Qh[((size_t)bh * T_ + t) * D_ + d] = hv;
      else if (region == 1) Kt[((size_t)bh * D_ + d) * T_ + t] = hv;
      else                  Vh[((size_t)bh * T_ + t) * D_ + d] = hv;
    }
  }
}

// out[8192x768] f32 = yh[8192x768] @ wo[768x768]
__global__ void out_gemm(const _Float16* __restrict__ yh,
                         const _Float16* __restrict__ wo,
                         float* __restrict__ out) {
  __shared__ _Float16 ldsB[2][BK][BPITCH];
  int wave = threadIdx.x >> 5;
  int lane = threadIdx.x & 31;
  int m0 = blockIdx.x * 128 + wave * 16;
  int n0 = blockIdx.y * 64;
  v8f acc[4] = {v8f{}, v8f{}, v8f{}, v8f{}};
  gemm_body(yh, C_, wo, C_, m0, n0, ldsB, acc);

  int mbase = (lane >> 4) * 8;
  int ncol  = lane & 15;
#pragma unroll
  for (int j = 0; j < 4; ++j)
#pragma unroll
    for (int g = 0; g < 8; ++g)
      out[(size_t)(m0 + g + mbase) * C_ + n0 + 16 * j + ncol] = acc[j][g];
}

// ---------------- flash attention: K/V tiles async-staged in LDS ------------
#define AW 4           // waves per block (each owns 16 queries)
#define KPITCH 40      // halves; Kt tile row pitch (80B)
#define VPITCH 72      // halves; V tile row pitch (144B)

__global__ void flash_attn(const _Float16* __restrict__ Qh,
                           const _Float16* __restrict__ Kt,
                           const _Float16* __restrict__ Vh,
                           _Float16* __restrict__ yh) {
  __shared__ _Float16 ldsK[2][64][KPITCH];   // [d][key] tile, 32 keys wide
  __shared__ _Float16 ldsV[2][32][VPITCH];   // [key][d] tile
  __shared__ _Float16 pt[AW][16][32];        // per-wave P staging

  int tid  = threadIdx.x;
  int wave = tid >> 5;
  int lane = tid & 31;
  int bh = blockIdx.y;
  int q0_blk = blockIdx.x * (AW * 16);
  int q0 = q0_blk + wave * 16;

  const _Float16* Qb  = Qh + (size_t)bh * T_ * D_;
  const _Float16* Ktb = Kt + (size_t)bh * D_ * T_;
  const _Float16* Vb  = Vh + (size_t)bh * T_ * D_;

  v16h aq0 = load_a_frag(Qb, D_, q0, 0);
  v16h aq1 = load_a_frag(Qb, D_, q0, 32);

  float mrow[8], ssum[8];
#pragma unroll
  for (int g = 0; g < 8; ++g) { mrow[g] = -__builtin_inff(); ssum[g] = 0.f; }
  v8f o[4] = {v8f{}, v8f{}, v8f{}, v8f{}};

  const float scale = 0.036084391824f;  // 1/sqrt(768)
  int mbase = (lane >> 4) * 8;
  int ncol  = lane & 15;
  int nkb = (q0_blk + AW * 16 - 1) / 32 + 1;   // uniform across block

  // copy-assignment for this thread: 2 chunks of K tile + 2 chunks of V tile
  // K tile: 64 rows x 64B; chunk id qk = 2*tid+s -> row qk>>2, col ((qk&3)*8)
  // V tile: 32 rows x 128B; chunk id qv = 2*tid+s -> row qv>>3, col ((qv&7)*8)
  auto issue_tile = [&](int buf, int key0) {
#pragma unroll
    for (int s = 0; s < 2; ++s) {
      int qk = 2 * tid + s;
      int kr = qk >> 2, kc = (qk & 3) * 8;
      async_b128_to_lds(Ktb + (size_t)kr * T_ + key0 + kc, &ldsK[buf][kr][kc]);
      int qv = 2 * tid + s;
      int vr = qv >> 3, vc = (qv & 7) * 8;
      async_b128_to_lds(Vb + (size_t)(key0 + vr) * D_ + vc, &ldsV[buf][vr][vc]);
    }
  };

  issue_tile(0, 0);
  for (int kb = 0; kb < nkb; ++kb) {
    int key0 = kb * 32;
    int more = (kb + 1 < nkb);
    if (more) issue_tile((kb + 1) & 1, key0 + 32);
    wait_async_le4(more);
    __syncthreads();

    if (key0 <= q0 + 15) {          // causal: this wave still has live keys
      const _Float16 (*kt)[KPITCH] = ldsK[kb & 1];
      const _Float16 (*vt)[VPITCH] = ldsV[kb & 1];

      v8f s0 = {}, s1 = {};
      s0 = wmma16(aq0, pack_b(&kt[0][0], KPITCH),   s0);
      s0 = wmma16(aq1, pack_b(&kt[32][0], KPITCH),  s0);
      s1 = wmma16(aq0, pack_b(&kt[0][16], KPITCH),  s1);
      s1 = wmma16(aq1, pack_b(&kt[32][16], KPITCH), s1);

      float alpha[8];
#pragma unroll
      for (int g = 0; g < 8; ++g) {
        int q = q0 + g + mbase;
        float v0 = s0[g] * scale;
        float v1 = s1[g] * scale;
        if (key0 + ncol      > q) v0 = -__builtin_inff();
        if (key0 + 16 + ncol > q) v1 = -__builtin_inff();
        float r = fmaxf(v0, v1);
#pragma unroll
        for (int off = 1; off < 16; off <<= 1) r = fmaxf(r, __shfl_xor(r, off, 16));
        float mn = fmaxf(mrow[g], r);
        float a  = __expf(mrow[g] - mn);
        mrow[g] = mn;
        float p0 = __expf(v0 - mn);
        float p1 = __expf(v1 - mn);
        s0[g] = p0; s1[g] = p1;
        float ps = p0 + p1;
#pragma unroll
        for (int off = 1; off < 16; off <<= 1) ps += __shfl_xor(ps, off, 16);
        ssum[g] = ssum[g] * a + ps;
        alpha[g] = a;
      }
#pragma unroll
      for (int c = 0; c < 4; ++c)
#pragma unroll
        for (int g = 0; g < 8; ++g) o[c][g] *= alpha[g];

      // reshape P: C-layout regs -> LDS -> A-layout frag (wave-private tile)
#pragma unroll
      for (int g = 0; g < 8; ++g) {
        int r = g + mbase;
        pt[wave][r][ncol]      = (_Float16)s0[g];
        pt[wave][r][16 + ncol] = (_Float16)s1[g];
      }
      asm volatile("s_wait_dscnt 0x0" ::: "memory");
      v16h pa;
      {
        const _Float16* pp = &pt[wave][ncol][mbase];
        v8h lo = *(const v8h*)pp;
        v8h hi = *(const v8h*)(pp + 16);
#pragma unroll
        for (int i = 0; i < 8; ++i) { pa[i] = lo[i]; pa[8 + i] = hi[i]; }
      }

#pragma unroll
      for (int c = 0; c < 4; ++c)
        o[c] = wmma16(pa, pack_b(&vt[0][16 * c], VPITCH), o[c]);
    }
    __syncthreads();
  }

  int b = bh / H_, h = bh % H_;
#pragma unroll
  for (int c = 0; c < 4; ++c)
#pragma unroll
    for (int g = 0; g < 8; ++g) {
      int q = q0 + g + mbase;
      float val = o[c][g] / ssum[g];
      yh[((size_t)(b * T_ + q)) * C_ + h * D_ + c * 16 + ncol] = (_Float16)val;
    }
}

extern "C" void kernel_launch(void* const* d_in, const int* in_sizes, int n_in,
                              void* d_out, int out_size, void* d_ws, size_t ws_size,
                              hipStream_t stream) {
  const float* x    = (const float*)d_in[0];   // [2,4096,768]
  const float* Wqkv = (const float*)d_in[1];   // [768,2304]
  const float* Wout = (const float*)d_in[2];   // [768,768]
  float* out = (float*)d_out;

  char* ws = (char*)d_ws;
  size_t off = 0;
  auto alloc = [&](size_t elems) {
    _Float16* p = (_Float16*)(ws + off);
    off += (elems * sizeof(_Float16) + 255) & ~(size_t)255;
    return p;
  };
  _Float16* xh  = alloc((size_t)BT_ * C_);
  _Float16* wqh = alloc((size_t)C_ * N3C_);
  _Float16* woh = alloc((size_t)C_ * C_);
  _Float16* Qh  = alloc((size_t)BT_ * C_);
  _Float16* Kt  = alloc((size_t)BT_ * C_);
  _Float16* Vh  = alloc((size_t)BT_ * C_);
  _Float16* yh  = alloc((size_t)BT_ * C_);

  {
    int n = BT_ * C_;
    cvt_f32_f16<<<(n + 255) / 256, 256, 0, stream>>>(x, xh, n);
    n = C_ * N3C_;
    cvt_f32_f16<<<(n + 255) / 256, 256, 0, stream>>>(Wqkv, wqh, n);
    n = C_ * C_;
    cvt_f32_f16<<<(n + 255) / 256, 256, 0, stream>>>(Wout, woh, n);
  }

  qkv_gemm<<<dim3(BT_ / 128, N3C_ / 64), 256, 0, stream>>>(xh, wqh, Qh, Kt, Vh);

  flash_attn<<<dim3((T_ / 16) / AW, B_ * H_), 32 * AW, 0, stream>>>(Qh, Kt, Vh, yh);

  out_gemm<<<dim3(BT_ / 128, C_ / 64), 256, 0, stream>>>(yh, woh, out);
}
